// EfficientHybridEncoder_79869211836816
// MI455X (gfx1250) — matrix operand order repaired and checked by reading, not compiled, not run on hardware
//
#include <hip/hip_runtime.h>
#include <hip/hip_bf16.h>

typedef _Float16 half_t;
typedef __attribute__((ext_vector_type(16))) _Float16 v16h;
typedef __attribute__((ext_vector_type(8)))  _Float16 v8h;
typedef __attribute__((ext_vector_type(8)))  float    v8f;

#define BATCH 4
#define DMODEL 256
#define NHEADS 8
#define HDIM 32
#define FFDIM 1024

// ---------------------------------------------------------------------------
// Generic WMMA GEMM: C[M,N] = A[M,K](f16) * Bt[Npad,K](f16, pre-transposed)
// Block: 256 threads (8 waves, 4 along M x 2 along N), tile 64x64, BK=64,
// double-buffered LDS (one barrier per chunk, loads overlap WMMA).
// M, K multiples of 64 (guaranteed); Npad = gridDim.y*64 (zero-padded).
// ---------------------------------------------------------------------------
struct EpiParams {
    const float* bias;      // [N] or null
    const float* resid;     // residual source or null
    float*       out_f32;   // optional f32 output (stride = logical N)
    half_t*      out_f16;   // optional f16 output (stride = logical N)
    int act;                // 0 none, 1 gelu(exact), 2 relu, 3 tanh
    int resid_mode;         // 0 none, 1 row-major [M,N], 2 NCHW feat [B,256,HW]
    int HW;                 // pixels per image (resid_mode 2 / remap)
    int remap;              // 1: out row = (m/HW)*tot_q + lvl_off + m%HW
    int lvl_off;
    int tot_q;
};

__global__ __launch_bounds__(256)
void k_gemm_f16(const half_t* __restrict__ A, const half_t* __restrict__ Bt,
                int M, int N, int K, EpiParams e)
{
    __shared__ __align__(16) half_t sA[2][64 * 64];   // [m][k]
    __shared__ __align__(16) half_t sB[2][64 * 64];   // [n][k]

    const int tid  = threadIdx.x;
    const int lane = tid & 31;
    const int wid  = tid >> 5;
    const int wm   = wid & 3;          // 4 x 16 rows
    const int wn   = wid >> 2;         // 2 x 32 cols
    const int row0 = blockIdx.x * 64;
    const int col0 = blockIdx.y * 64;
    const int lg   = lane >> 4;
    const int ll   = lane & 15;

    // staging: thread -> 16 halves (2 x b128) per matrix
    const int sr = tid >> 2;           // tile row 0..63
    const int sk = (tid & 3) * 16;     // k base 0/16/32/48

    const half_t* gA = A  + (size_t)(row0 + sr) * K + sk;
    const half_t* gB = Bt + (size_t)(col0 + sr) * K + sk;
    const int sOff = sr * 64 + sk;

    v8f acc0 = {}; v8f acc1 = {};

    // compute one 64-wide K chunk resident in buffer (cA, cB)
    auto compute = [&](const half_t* cA, const half_t* cB) {
        #pragma unroll
        for (int s = 0; s < 2; ++s) {
            v16h fa, fb0, fb1;
            {   // A frag: halves -> K = lg*16..+15
                const half_t* p = cA + (wm * 16 + ll) * 64 + s * 32 + lg * 16;
                v8h lo = *(const v8h*)p;
                v8h hi = *(const v8h*)(p + 8);
                #pragma unroll
                for (int i = 0; i < 8; ++i) { fa[i] = lo[i]; fa[8 + i] = hi[i]; }
            }
            {   // B frag 0: column n = wn*32 + ll, halves -> K = lg*16..+15
                const half_t* p = cB + (wn * 32 + ll) * 64 + s * 32 + lg * 16;
                v8h lo = *(const v8h*)p;
                v8h hi = *(const v8h*)(p + 8);
                #pragma unroll
                for (int i = 0; i < 8; ++i) { fb0[i] = lo[i]; fb0[8 + i] = hi[i]; }
            }
            {   // B frag 1: column n + 16
                const half_t* p = cB + (wn * 32 + 16 + ll) * 64 + s * 32 + lg * 16;
                v8h lo = *(const v8h*)p;
                v8h hi = *(const v8h*)(p + 8);
                #pragma unroll
                for (int i = 0; i < 8; ++i) { fb1[i] = lo[i]; fb1[8 + i] = hi[i]; }
            }
            acc0 = __builtin_amdgcn_wmma_f32_16x16x32_f16(false, fa, false, fb0,
                                                          (short)0, acc0, false, false);
            acc1 = __builtin_amdgcn_wmma_f32_16x16x32_f16(false, fa, false, fb1,
                                                          (short)0, acc1, false, false);
        }
    };

    const int nchunks = K >> 6;

    // prologue: stage chunk 0 into buffer 0
    *(v8h*)(sA[0] + sOff)     = *(const v8h*)gA;
    *(v8h*)(sA[0] + sOff + 8) = *(const v8h*)(gA + 8);
    *(v8h*)(sB[0] + sOff)     = *(const v8h*)gB;
    *(v8h*)(sB[0] + sOff + 8) = *(const v8h*)(gB + 8);
    __syncthreads();

    // steady state: prefetch chunk c+1 while computing chunk c
    for (int c = 0; c < nchunks - 1; ++c) {
        const int cur = c & 1;
        const int k1 = (c + 1) << 6;
        // 1) issue next-chunk global loads (latency hidden behind WMMAs)
        const v8h na0 = *(const v8h*)(gA + k1);
        const v8h na1 = *(const v8h*)(gA + k1 + 8);
        const v8h nb0 = *(const v8h*)(gB + k1);
        const v8h nb1 = *(const v8h*)(gB + k1 + 8);
        // 2) compute on resident buffer
        compute(sA[cur], sB[cur]);
        // 3) store prefetched data into alternate buffer
        half_t* dA = sA[cur ^ 1];
        half_t* dB = sB[cur ^ 1];
        *(v8h*)(dA + sOff)     = na0;
        *(v8h*)(dA + sOff + 8) = na1;
        *(v8h*)(dB + sOff)     = nb0;
        *(v8h*)(dB + sOff + 8) = nb1;
        __syncthreads();
    }
    // epilogue chunk
    compute(sA[(nchunks - 1) & 1], sB[(nchunks - 1) & 1]);

    // ---- epilogue (logical-N strides; clamp cols for padded GEMMs) ----
    const int rbase = row0 + wm * 16 + lg * 8;
    #pragma unroll
    for (int j = 0; j < 2; ++j) {
        const v8f acc = j ? acc1 : acc0;
        const int col = col0 + wn * 32 + j * 16 + ll;
        if (col >= N) continue;
        const float bb = e.bias ? e.bias[col] : 0.f;
        #pragma unroll
        for (int r = 0; r < 8; ++r) {
            const int row = rbase + r;
            float v = acc[r] + bb;
            if (e.act == 1)      v = 0.5f * v * (1.f + erff(v * 0.7071067811865475f));
            else if (e.act == 2) v = fmaxf(v, 0.f);
            else if (e.act == 3) v = tanhf(v);
            if (e.resid_mode == 1) {
                v += e.resid[(size_t)row * N + col];
            } else if (e.resid_mode == 2) {
                const int bidx = row / e.HW, hw = row % e.HW;
                v += e.resid[((size_t)bidx * DMODEL + col) * e.HW + hw];
            }
            if (e.out_f32) {
                size_t orow = row;
                if (e.remap)
                    orow = (size_t)(row / e.HW) * e.tot_q + e.lvl_off + (row % e.HW);
                e.out_f32[orow * (size_t)N + col] = v;
            }
            if (e.out_f16)
                e.out_f16[(size_t)row * (size_t)N + col] = (half_t)v;
        }
    }
}

// ---------------------------------------------------------------------------
// Weight conversion kernels
// ---------------------------------------------------------------------------
__global__ void k_f32_to_f16(const float* __restrict__ src,
                             half_t* __restrict__ dst, int n)
{
    int i = blockIdx.x * blockDim.x + threadIdx.x;
    if (i < n) dst[i] = (half_t)src[i];
}

// src [K,N] row-major  ->  dst [Npad,K] (B transposed, zero-filled pad rows)
__global__ void k_w_to_bt(const float* __restrict__ src, half_t* __restrict__ dst,
                          int K, int N, int Npad)
{
    int i = blockIdx.x * blockDim.x + threadIdx.x;
    if (i >= Npad * K) return;
    const int n = i / K, k = i % K;
    dst[i] = (n < N) ? (half_t)src[(size_t)k * N + n] : (half_t)0.f;
}

// ---------------------------------------------------------------------------
// Depthwise 3x3 conv + bias -> token-major f16 [(b*HW+hw), D]
// ---------------------------------------------------------------------------
__global__ void k_dwconv(const float* __restrict__ x, const float* __restrict__ w,
                         const float* __restrict__ bias, half_t* __restrict__ out,
                         int Hh, int Ww)
{
    const int HW = Hh * Ww;
    int idx = blockIdx.x * blockDim.x + threadIdx.x;
    if (idx >= BATCH * HW * DMODEL) return;
    const int c  = idx % DMODEL;
    const int hw = (idx / DMODEL) % HW;
    const int b  = idx / (DMODEL * HW);
    const int h = hw / Ww, wc = hw % Ww;
    const float* xp = x + ((size_t)b * DMODEL + c) * HW;
    const float* wp = w + c * 9;
    float acc = bias[c];
    #pragma unroll
    for (int dy = -1; dy <= 1; ++dy)
        #pragma unroll
        for (int dx = -1; dx <= 1; ++dx) {
            int hh = h + dy, ww2 = wc + dx;
            if (hh >= 0 && hh < Hh && ww2 >= 0 && ww2 < Ww)
                acc += xp[hh * Ww + ww2] * wp[(dy + 1) * 3 + (dx + 1)];
        }
    out[idx] = (half_t)acc;
}

// ---------------------------------------------------------------------------
// LayerNorm (one block of 256 threads per token) + optional sincos pos embed
// ---------------------------------------------------------------------------
__global__ __launch_bounds__(256)
void k_layernorm(const float* __restrict__ x, const float* __restrict__ g,
                 const float* __restrict__ bta, half_t* __restrict__ out,
                 int addpos, int Hh, int Ww, const float* __restrict__ lvl)
{
    __shared__ float red[256];
    const int m = blockIdx.x, t = threadIdx.x;
    const float v = x[(size_t)m * DMODEL + t];
    red[t] = v; __syncthreads();
    for (int s = 128; s > 0; s >>= 1) { if (t < s) red[t] += red[t + s]; __syncthreads(); }
    const float mean = red[0] * (1.f / DMODEL); __syncthreads();
    const float d = v - mean;
    red[t] = d * d; __syncthreads();
    for (int s = 128; s > 0; s >>= 1) { if (t < s) red[t] += red[t + s]; __syncthreads(); }
    const float var = red[0] * (1.f / DMODEL);
    float y = d * rsqrtf(var + 1e-5f) * g[t] + bta[t];
    if (addpos) {
        const int HW = Hh * Ww;
        const int hw = m % HW;
        const float gh = (float)(hw / Ww), gw = (float)(hw % Ww);
        const int q = t & 63;
        const float denom = __powf(10000.f, (float)q * (1.f / 64.f));
        const float ang = ((t < 128) ? gh : gw) / denom;
        const float pe = ((t >> 6) & 1) ? __cosf(ang) : __sinf(ang);
        y += pe + lvl[t];
    }
    out[(size_t)m * DMODEL + t] = (half_t)y;
}

// ---------------------------------------------------------------------------
// Softmax over 12 per (token, head); in-place + mirror into d_out attn block
// ---------------------------------------------------------------------------
__global__ void k_softmax12(float* __restrict__ aw, float* __restrict__ attn_out,
                            int total)
{
    int i = blockIdx.x * blockDim.x + threadIdx.x;
    if (i >= total) return;
    float* p = aw + (size_t)i * 12;
    float mx = -1e30f;
    #pragma unroll
    for (int j = 0; j < 12; ++j) mx = fmaxf(mx, p[j]);
    float e[12]; float s = 0.f;
    #pragma unroll
    for (int j = 0; j < 12; ++j) { e[j] = __expf(p[j] - mx); s += e[j]; }
    const float inv = 1.f / s;
    #pragma unroll
    for (int j = 0; j < 12; ++j) {
        const float v = e[j] * inv;
        p[j] = v;
        attn_out[(size_t)i * 12 + j] = v;
    }
}

// ---------------------------------------------------------------------------
// Deformable sampling: one wave per (token, head); lane = head-channel.
// ---------------------------------------------------------------------------
__global__ __launch_bounds__(256)
void k_sample(const float* __restrict__ off, const float* __restrict__ aw,
              const float* __restrict__ t0, const float* __restrict__ t1,
              const float* __restrict__ t2, half_t* __restrict__ aggh,
              int Qcur, int Hc, int Wc)
{
    const int gwv  = blockIdx.x * 8 + (threadIdx.x >> 5);
    const int lane = threadIdx.x & 31;
    if (gwv >= BATCH * Qcur * NHEADS) return;
    const int head = gwv % NHEADS;
    const int m    = gwv / NHEADS;         // b*Q + q
    const int b    = m / Qcur;
    const int q    = m % Qcur;
    const float rx = (((q % Wc) + 0.5f) / Wc) * 2.f - 1.f;
    const float ry = (((q / Wc) + 0.5f) / Hc) * 2.f - 1.f;
    const float* offp = off + (size_t)m * 192 + head * 24;   // [lvl][p][xy]
    const float* awp  = aw  + (size_t)m * 96  + head * 12;   // [lvl][p]
    const float* fl[3] = { t0, t1, t2 };
    const int HS[3] = { 80, 40, 20 }, WS[3] = { 80, 40, 20 };

    float acc = 0.f;
    #pragma unroll
    for (int lvl = 0; lvl < 3; ++lvl) {
        const float* f = fl[lvl];
        const int Hsz = HS[lvl], Wsz = WS[lvl];
        #pragma unroll
        for (int p = 0; p < 4; ++p) {
            const float gx = rx + offp[lvl * 8 + p * 2 + 0];
            const float gy = ry + offp[lvl * 8 + p * 2 + 1];
            const float w  = awp[lvl * 4 + p];
            const float xx = (gx + 1.f) * 0.5f * (Wsz - 1);
            const float yy = (gy + 1.f) * 0.5f * (Hsz - 1);
            const float x0f = floorf(xx), y0f = floorf(yy);
            const int x0 = (int)x0f, y0 = (int)y0f;
            const float wx1 = xx - x0f, wy1 = yy - y0f;
            const float wx0 = 1.f - wx1, wy0 = 1.f - wy1;
            float sv = 0.f;
            #pragma unroll
            for (int cy = 0; cy < 2; ++cy)
                #pragma unroll
                for (int cx = 0; cx < 2; ++cx) {
                    const int xi = x0 + cx, yi = y0 + cy;
                    if (xi >= 0 && xi < Wsz && yi >= 0 && yi < Hsz) {
                        const float cw = (cx ? wx1 : wx0) * (cy ? wy1 : wy0);
                        sv += cw * f[((size_t)(b * Hsz * Wsz + yi * Wsz + xi)) * DMODEL
                                     + head * HDIM + lane];
                    }
                }
            acc += w * sv;
        }
    }
    aggh[(size_t)m * DMODEL + head * HDIM + lane] = (half_t)acc;
}

// ---------------------------------------------------------------------------
// Host-side orchestration
// ---------------------------------------------------------------------------
extern "C" void kernel_launch(void* const* d_in, const int* in_sizes, int n_in,
                              void* d_out, int out_size, void* d_ws, size_t ws_size,
                              hipStream_t stream)
{
    const float* feat[3] = { (const float*)d_in[0], (const float*)d_in[1],
                             (const float*)d_in[2] };
    const float* dw_w  = (const float*)d_in[3];
    const float* dw_b  = (const float*)d_in[4];
    const float* pw_w  = (const float*)d_in[5];
    const float* pw_b  = (const float*)d_in[6];
    const float* off_w = (const float*)d_in[7];
    const float* off_b = (const float*)d_in[8];
    const float* aw_w  = (const float*)d_in[9];
    const float* aw_b  = (const float*)d_in[10];
    const float* op_w  = (const float*)d_in[11];
    const float* op_b  = (const float*)d_in[12];
    const float* nq_g  = (const float*)d_in[13];
    const float* nq_b  = (const float*)d_in[14];
    const float* nf_g  = (const float*)d_in[15];
    const float* nf_b  = (const float*)d_in[16];
    const float* ff1_w = (const float*)d_in[17];
    const float* ff1_b = (const float*)d_in[18];
    const float* ff2_w = (const float*)d_in[19];
    const float* ff2_b = (const float*)d_in[20];
    const float* lvl_e = (const float*)d_in[21];

    const int Hs[3] = { 80, 40, 20 }, Ws[3] = { 80, 40, 20 };
    const int HW[3] = { 6400, 1600, 400 };
    const int Ml[3] = { BATCH * 6400, BATCH * 1600, BATCH * 400 };
    const int lvl_off[3] = { 0, 6400, 8000 };

    // ---- scratch layout (bump allocator over d_ws) ----
    char* ws = (char*)d_ws;
    size_t o = 0;
    auto alloc = [&](size_t bytes) -> char* {
        char* p = ws + o;
        o += (bytes + 255) & ~(size_t)255;
        return p;
    };
    // weights (Bt layout [Npad, K], f16). pw_w is (o,i) == [N,K] already.
    half_t* pw16   = (half_t*)alloc((size_t)3 * 256 * 256 * 2);
    half_t* offBt  = (half_t*)alloc((size_t)192 * 256 * 2);
    half_t* awBt   = (half_t*)alloc((size_t)128 * 256 * 2);   // N=96 padded to 128
    half_t* opBt   = (half_t*)alloc((size_t)256 * 256 * 2);
    half_t* ff1Bt  = (half_t*)alloc((size_t)1024 * 256 * 2);
    half_t* ff2Bt  = (half_t*)alloc((size_t)256 * 1024 * 2);
    float* tok[3];
    for (int l = 0; l < 3; ++l) tok[l] = (float*)alloc((size_t)Ml[l] * 256 * 4);
    const int M0 = Ml[0];
    half_t* bufA16 = (half_t*)alloc((size_t)M0 * 256 * 2);    // dwconv out / agg
    half_t* bufQ16 = (half_t*)alloc((size_t)M0 * 256 * 2);    // q / h
    float*  bufOff = (float*)alloc((size_t)M0 * 192 * 4);
    float*  bufAw  = (float*)alloc((size_t)M0 * 96 * 4);
    float*  tok2   = (float*)alloc((size_t)M0 * 256 * 4);
    half_t* bufFF  = (half_t*)alloc((size_t)M0 * 1024 * 2);

    // ---- weight conversions ----
    {
        int n = 3 * 256 * 256;
        k_f32_to_f16<<<(n + 255) / 256, 256, 0, stream>>>(pw_w, pw16, n);
        n = 192 * 256;
        k_w_to_bt<<<(n + 255) / 256, 256, 0, stream>>>(off_w, offBt, 256, 192, 192);
        n = 128 * 256;
        k_w_to_bt<<<(n + 255) / 256, 256, 0, stream>>>(aw_w, awBt, 256, 96, 128);
        n = 256 * 256;
        k_w_to_bt<<<(n + 255) / 256, 256, 0, stream>>>(op_w, opBt, 256, 256, 256);
        n = 1024 * 256;
        k_w_to_bt<<<(n + 255) / 256, 256, 0, stream>>>(ff1_w, ff1Bt, 256, 1024, 1024);
        n = 256 * 1024;
        k_w_to_bt<<<(n + 255) / 256, 256, 0, stream>>>(ff2_w, ff2Bt, 1024, 256, 256);
    }

    // ---- stage 1: local branch (all levels, feats_local needed by sampling) ----
    for (int l = 0; l < 3; ++l) {
        const int total = Ml[l] * 256;
        k_dwconv<<<(total + 255) / 256, 256, 0, stream>>>(
            feat[l], dw_w + (size_t)l * 256 * 9, dw_b + l * 256, bufA16, Hs[l], Ws[l]);
        EpiParams e{};
        e.bias = pw_b + l * 256; e.act = 1;
        e.resid = feat[l]; e.resid_mode = 2; e.HW = HW[l];
        e.out_f32 = tok[l];
        dim3 g(Ml[l] / 64, 4);
        k_gemm_f16<<<g, 256, 0, stream>>>(bufA16, pw16 + (size_t)l * 256 * 256,
                                          Ml[l], 256, 256, e);
    }

    float* out_f = (float*)d_out;
    float* attn_base = out_f + (size_t)BATCH * 8400 * 256;
    size_t attn_off = 0;

    // ---- stage 2: per-level deformable attention + FFN ----
    for (int l = 0; l < 3; ++l) {
        const int M = Ml[l];
        k_layernorm<<<M, 256, 0, stream>>>(tok[l], nq_g, nq_b, bufQ16,
                                           1, Hs[l], Ws[l], lvl_e + l * 256);
        { EpiParams e{}; e.bias = off_b; e.act = 3; e.out_f32 = bufOff;
          dim3 g(M / 64, 3);
          k_gemm_f16<<<g, 256, 0, stream>>>(bufQ16, offBt, M, 192, 256, e); }
        { EpiParams e{}; e.bias = aw_b; e.out_f32 = bufAw;
          dim3 g(M / 64, 2);   // Npad = 128, epilogue clamps to N = 96
          k_gemm_f16<<<g, 256, 0, stream>>>(bufQ16, awBt, M, 96, 256, e); }
        const int tot = M * NHEADS;
        k_softmax12<<<(tot + 255) / 256, 256, 0, stream>>>(bufAw, attn_base + attn_off, tot);
        attn_off += (size_t)tot * 12;
        k_sample<<<(tot + 7) / 8, 256, 0, stream>>>(bufOff, bufAw, tok[0], tok[1],
                                                    tok[2], bufA16, HW[l], Hs[l], Ws[l]);
        { EpiParams e{}; e.bias = op_b; e.resid = tok[l]; e.resid_mode = 1;
          e.out_f32 = tok2;
          dim3 g(M / 64, 4);
          k_gemm_f16<<<g, 256, 0, stream>>>(bufA16, opBt, M, 256, 256, e); }
        k_layernorm<<<M, 256, 0, stream>>>(tok2, nf_g, nf_b, bufQ16,
                                           0, Hs[l], Ws[l], nullptr);
        { EpiParams e{}; e.bias = ff1_b; e.act = 2; e.out_f16 = bufFF;
          dim3 g(M / 64, 16);
          k_gemm_f16<<<g, 256, 0, stream>>>(bufQ16, ff1Bt, M, 1024, 256, e); }
        { EpiParams e{}; e.bias = ff2_b; e.resid = tok2; e.resid_mode = 1;
          e.out_f32 = out_f; e.remap = 1; e.HW = HW[l];
          e.lvl_off = lvl_off[l]; e.tot_q = 8400;
          dim3 g(M / 64, 4);
          k_gemm_f16<<<g, 256, 0, stream>>>(bufFF, ff2Bt, M, 256, 1024, e); }
    }
}
